// SA_MH_Layer_77343771066719
// MI455X (gfx1250) — matrix operand order repaired and checked by reading, not compiled, not run on hardware
//
#include <hip/hip_runtime.h>
#include <hip/hip_bf16.h>

// ---------------------------------------------------------------------------
// Problem constants (match reference)
// ---------------------------------------------------------------------------
constexpr int NB = 4;      // batch
constexpr int NC = 256;    // channels
constexpr int NP = 4096;   // points
constexpr int NH = 4;      // heads
constexpr int HD = 64;     // head dim
constexpr float BN_EPS = 1e-5f;

typedef __attribute__((ext_vector_type(16))) _Float16 v16h;
typedef __attribute__((ext_vector_type(8)))  float    v8f;

// K-offset inside a 16-bit A/B WMMA fragment (16x32), per CDNA5 ISA 7.12.2:
// VGPR j (j<4): K = 2j   (+8 for lanes 16-31)
// VGPR j (j>=4): K = 16+2(j-4) (+8 for lanes 16-31)
__device__ __forceinline__ int frag_k(int j, int hl) {
    return 2 * (j & 3) + ((j >> 2) << 4) + (hl << 3);
}

__device__ __forceinline__ v8f wmma_f16(const v16h& a, const v16h& b, const v8f& c) {
    return __builtin_amdgcn_wmma_f32_16x16x32_f16(false, a, false, b, (short)0, c,
                                                  false, false);
}

// CDNA5 async global->LDS copy (ASYNCcnt path).  VFLAT encoding: VDST = LDS
// byte address VGPR, VADDR = 64-bit global address VGPR pair.
__device__ __forceinline__ void async_copy_b128(unsigned lds_byte, const void* gptr) {
    unsigned long long ga = (unsigned long long)gptr;
    asm volatile("global_load_async_to_lds_b128 %0, %1, off"
                 :: "v"(lds_byte), "v"(ga)
                 : "memory");
}

__device__ __forceinline__ void wait_async0() {
#if __has_builtin(__builtin_amdgcn_s_wait_asynccnt)
    __builtin_amdgcn_s_wait_asynccnt(0);
#else
    asm volatile("s_wait_asynccnt 0" ::: "memory");
#endif
}

// ---------------------------------------------------------------------------
// Kernel 0: zero BN stat accumulators
// ---------------------------------------------------------------------------
__global__ void zero_stats_kernel(float* __restrict__ gsum, float* __restrict__ gsq) {
    int i = blockIdx.x * blockDim.x + threadIdx.x;
    if (i < NC) { gsum[i] = 0.0f; gsq[i] = 0.0f; }
}

// ---------------------------------------------------------------------------
// Kernel 1a: convert the four weight matrices to f16 (done once per launch)
// ---------------------------------------------------------------------------
__global__ __launch_bounds__(256)
void cvt_weights_kernel(const float* __restrict__ wq, const float* __restrict__ wk,
                        const float* __restrict__ wv, const float* __restrict__ wt,
                        _Float16* __restrict__ wqh, _Float16* __restrict__ wkh,
                        _Float16* __restrict__ wvh, _Float16* __restrict__ wth) {
    int i = blockIdx.x * 256 + threadIdx.x;
    if (i < NC * NC) {
        wqh[i] = (_Float16)wq[i];
        wkh[i] = (_Float16)wk[i];
        wvh[i] = (_Float16)wv[i];
        wth[i] = (_Float16)wt[i];
    }
}

// Kernel 1b: convert x to f16 [b, c, p]
__global__ __launch_bounds__(256)
void cvt_x_kernel(const float* __restrict__ x, _Float16* __restrict__ xh) {
    int i = blockIdx.x * 256 + threadIdx.x;
    if (i < NB * NC * NP) xh[i] = (_Float16)x[i];
}

// ---------------------------------------------------------------------------
// Kernel 2: QKV projection via WMMA, activations staged through LDS with
// async global->LDS copies.
//   Output layouts (f16):
//     qT, kT : [b, h, p, d]
//     vv     : [b, c, p]
// Grid: (NP/128, NC/16, NB), block 256 threads = 8 waves, each wave one 16x16 tile.
// ---------------------------------------------------------------------------
__global__ __launch_bounds__(256)
void proj_qkv_kernel(const _Float16* __restrict__ xh,
                     const _Float16* __restrict__ wqh,
                     const _Float16* __restrict__ wkh,
                     const _Float16* __restrict__ wvh,
                     _Float16* __restrict__ qT,
                     _Float16* __restrict__ kT,
                     _Float16* __restrict__ vv) {
    __shared__ _Float16 sX[32 * 128];   // 32 input channels x 128 points (8 KB)

    const int tid  = threadIdx.x;
    const int lane = tid & 31;
    const int wave = tid >> 5;
    const int hl   = lane >> 4;
    const int lq   = lane & 15;

    const int pblk = blockIdx.x * 128;
    const int p0   = pblk + wave * 16;
    const int m0   = blockIdx.y * 16;
    const int b    = blockIdx.z;

    const _Float16* __restrict__ xb = xh + (size_t)b * NC * NP;

    // per-thread async staging slot: thread copies 16 f16 (32 bytes)
    const int r    = tid >> 3;          // tile row (input channel) 0..31
    const int colb = (tid & 7) * 16;    // tile col base 0..112
    const unsigned ldsDst =
        (unsigned)(unsigned long long)(void*)&sX[r * 128 + colb];

    v8f cq = {}, ck = {}, cv = {};
    for (int kt = 0; kt < NC / 32; ++kt) {
        const int kbase = kt * 32;
        // ---- stage x tile [32 cin][128 p] into LDS (async) ----------------
        async_copy_b128(ldsDst, xb + (size_t)(kbase + r) * NP + pblk + colb);
        async_copy_b128(ldsDst + 16, xb + (size_t)(kbase + r) * NP + pblk + colb + 8);
        wait_async0();
        __syncthreads();

        v16h aq, ak, av, bx;
#pragma unroll
        for (int j = 0; j < 8; ++j) {
            const int kk = frag_k(j, hl);
            const int wrow = (m0 + lq) * NC + kbase + kk;
            aq[2 * j]     = wqh[wrow];
            aq[2 * j + 1] = wqh[wrow + 1];
            ak[2 * j]     = wkh[wrow];
            ak[2 * j + 1] = wkh[wrow + 1];
            av[2 * j]     = wvh[wrow];
            av[2 * j + 1] = wvh[wrow + 1];
            bx[2 * j]     = sX[kk * 128 + wave * 16 + lq];
            bx[2 * j + 1] = sX[(kk + 1) * 128 + wave * 16 + lq];
        }
        cq = wmma_f16(aq, bx, cq);
        ck = wmma_f16(ak, bx, ck);
        cv = wmma_f16(av, bx, cv);
        __syncthreads();   // sX reused next K step
    }

#pragma unroll
    for (int j = 0; j < 8; ++j) {
        const int c  = m0 + j + 8 * hl;  // out channel (D row)
        const int p  = p0 + lq;          // point (D col)
        const int hh = c >> 6;           // head
        const int dd = c & 63;           // dim within head
        const size_t qoff = (((size_t)b * NH + hh) * NP + p) * (size_t)HD + dd;
        qT[qoff] = (_Float16)cq[j];
        kT[qoff] = (_Float16)ck[j];
        vv[((size_t)b * NC + c) * NP + p] = (_Float16)cv[j];
    }
}

// ---------------------------------------------------------------------------
// Kernel 3: flash attention.  One wave per (b, h, 16-query tile).
//   S = q^T k / 8, online softmax over keys (32 per step), O = P @ v^T.
//   Epilogue writes dh = O - x  (f16, [b,c,p]) for the trans-conv GEMM.
// ---------------------------------------------------------------------------
__global__ __launch_bounds__(32)
void flash_attn_kernel(const _Float16* __restrict__ qT,
                       const _Float16* __restrict__ kT,
                       const _Float16* __restrict__ vv,
                       const float* __restrict__ x,
                       _Float16* __restrict__ dh) {
    __shared__ float sS[16][33];   // score tile: 16 queries x 32 keys, padded
    __shared__ float sFac[16];     // per-row rescale factors / inverse sums

    const int lane = threadIdx.x & 31;
    const int hl   = lane >> 4;
    const int lq   = lane & 15;

    const int itile = blockIdx.x;       // query tile
    const int hh    = blockIdx.y;       // head
    const int b     = blockIdx.z;
    const int bh    = b * NH + hh;
    const int pi    = itile * 16;

    const _Float16* __restrict__ qb = qT + (size_t)bh * NP * HD;
    const _Float16* __restrict__ kb = kT + (size_t)bh * NP * HD;
    const _Float16* __restrict__ vb = vv + ((size_t)b * NC + hh * HD) * NP;

    // A fragments of q^T (M=query row, K=d), two K-steps of 32 covering d=64
    v16h aq0, aq1;
    {
        const _Float16* qrow = qb + (size_t)(pi + lq) * HD;
#pragma unroll
        for (int j = 0; j < 8; ++j) {
            const int kk  = frag_k(j, hl);
            aq0[2 * j]     = qrow[kk];
            aq0[2 * j + 1] = qrow[kk + 1];
            aq1[2 * j]     = qrow[32 + kk];
            aq1[2 * j + 1] = qrow[32 + kk + 1];
        }
    }

    float row_max = -3.0e38f;   // valid in lanes 0..15 (one query row each)
    float row_sum = 0.0f;
    v8f co[4] = {{}, {}, {}, {}};   // O accumulators: 16 queries x 64 dims

    for (int jt = 0; jt < NP / 32; ++jt) {
        const int pj = jt * 32;

        // prefetch next K/V tiles into cache (global_prefetch_b8)
        if (jt + 1 < NP / 32) {
            __builtin_prefetch(kb + (size_t)(pj + 32 + lane) * HD, 0, 0);
            __builtin_prefetch(vb + (size_t)(lane & 15) * NP + pj + 32, 0, 0);
        }

        // ---- scores for two 16-key subtiles -------------------------------
#pragma unroll
        for (int js = 0; js < 2; ++js) {
            v16h bk0, bk1;
            const _Float16* krow = kb + (size_t)(pj + js * 16 + lq) * HD;
#pragma unroll
            for (int j = 0; j < 8; ++j) {
                const int kk  = frag_k(j, hl);
                bk0[2 * j]     = krow[kk];
                bk0[2 * j + 1] = krow[kk + 1];
                bk1[2 * j]     = krow[32 + kk];
                bk1[2 * j + 1] = krow[32 + kk + 1];
            }
            v8f s = {};
            s = wmma_f16(aq0, bk0, s);
            s = wmma_f16(aq1, bk1, s);
#pragma unroll
            for (int j = 0; j < 8; ++j)
                sS[j + 8 * hl][js * 16 + lq] = s[j] * 0.125f;  // 1/sqrt(64)
        }
        __syncthreads();

        // ---- online softmax over the 32 new keys (one lane per query row) -
        if (lane < 16) {
            float mx = row_max;
#pragma unroll
            for (int c = 0; c < 32; ++c) mx = fmaxf(mx, sS[lane][c]);
            const float fac = __expf(row_max - mx);
            float se = 0.0f;
#pragma unroll
            for (int c = 0; c < 32; ++c) {
                const float e = __expf(sS[lane][c] - mx);
                sS[lane][c] = e;
                se += e;
            }
            row_sum = row_sum * fac + se;
            row_max = mx;
            sFac[lane] = fac;
        }
        __syncthreads();

        // ---- rescale O accumulators, build P fragment (A-layout, K=32 keys)
        v16h ap;
#pragma unroll
        for (int j = 0; j < 8; ++j) {
            const float f = sFac[j + 8 * hl];
            co[0][j] *= f; co[1][j] *= f; co[2][j] *= f; co[3][j] *= f;
            const int kk = frag_k(j, hl);
            ap[2 * j]     = (_Float16)sS[lq][kk];
            ap[2 * j + 1] = (_Float16)sS[lq][kk + 1];
        }

        // ---- O += P @ v^T  (4 d-subtiles of 16) ---------------------------
#pragma unroll
        for (int t = 0; t < 4; ++t) {
            v16h bvf;
            const _Float16* vrow = vb + (size_t)(t * 16 + lq) * NP + pj;
#pragma unroll
            for (int j = 0; j < 8; ++j) {
                const int kk = frag_k(j, hl);
                bvf[2 * j]     = vrow[kk];
                bvf[2 * j + 1] = vrow[kk + 1];
            }
            co[t] = wmma_f16(ap, bvf, co[t]);
        }
        __syncthreads();   // sS reused next iteration
    }

    // ---- finalize: divide by row sums, store dh = o - x (f16) -------------
    if (lane < 16) sFac[lane] = 1.0f / row_sum;
    __syncthreads();
#pragma unroll
    for (int t = 0; t < 4; ++t) {
        const int c = hh * HD + t * 16 + lq;      // D col (n) = head dim
#pragma unroll
        for (int j = 0; j < 8; ++j) {
            const float inv = sFac[j + 8 * hl];
            const int p = pi + j + 8 * hl;        // D row (m) = query pos
            const size_t off = ((size_t)b * NC + c) * NP + p;
            dh[off] = (_Float16)(co[t][j] * inv - x[off]);
        }
    }
}

// ---------------------------------------------------------------------------
// Kernel 4: y = Wt @ dh + bt ; store y (f32) and accumulate per-channel
//           sum / sum-of-squares for BatchNorm.  dh staged via async LDS copy.
// Grid: (NP/128, NC/16, NB), block 256 = 8 waves.
// ---------------------------------------------------------------------------
__global__ __launch_bounds__(256)
void trans_conv_kernel(const _Float16* __restrict__ dh,
                       const _Float16* __restrict__ wth,
                       const float* __restrict__ bt,
                       float* __restrict__ yy,
                       float* __restrict__ gsum,
                       float* __restrict__ gsq) {
    __shared__ _Float16 sX[32 * 128];
    __shared__ float red[8][16][17];

    const int tid  = threadIdx.x;
    const int lane = tid & 31;
    const int wave = tid >> 5;
    const int hl   = lane >> 4;
    const int lq   = lane & 15;

    const int pblk = blockIdx.x * 128;
    const int p0   = pblk + wave * 16;
    const int m0   = blockIdx.y * 16;
    const int b    = blockIdx.z;

    const _Float16* __restrict__ db = dh + (size_t)b * NC * NP;

    const int r    = tid >> 3;
    const int colb = (tid & 7) * 16;
    const unsigned ldsDst =
        (unsigned)(unsigned long long)(void*)&sX[r * 128 + colb];

    v8f acc = {};
    for (int kt = 0; kt < NC / 32; ++kt) {
        const int kbase = kt * 32;
        async_copy_b128(ldsDst, db + (size_t)(kbase + r) * NP + pblk + colb);
        async_copy_b128(ldsDst + 16, db + (size_t)(kbase + r) * NP + pblk + colb + 8);
        wait_async0();
        __syncthreads();

        v16h aw, bo;
#pragma unroll
        for (int j = 0; j < 8; ++j) {
            const int kk = frag_k(j, hl);
            const int wrow = (m0 + lq) * NC + kbase + kk;
            aw[2 * j]     = wth[wrow];
            aw[2 * j + 1] = wth[wrow + 1];
            bo[2 * j]     = sX[kk * 128 + wave * 16 + lq];
            bo[2 * j + 1] = sX[(kk + 1) * 128 + wave * 16 + lq];
        }
        acc = wmma_f16(aw, bo, acc);
        __syncthreads();
    }

#pragma unroll
    for (int j = 0; j < 8; ++j) {
        const int c = m0 + j + 8 * hl;
        const int p = p0 + lq;
        const float yv = acc[j] + bt[c];
        yy[((size_t)b * NC + c) * NP + p] = yv;
        red[wave][j + 8 * hl][lq] = yv;
    }
    __syncthreads();

    if (tid < 128) {
        const int w = tid >> 4;
        const int rr = tid & 15;
        float s = 0.0f, s2 = 0.0f;
#pragma unroll
        for (int n = 0; n < 16; ++n) {
            const float v = red[w][rr][n];
            s  += v;
            s2 += v * v;
        }
        atomicAdd(&gsum[m0 + rr], s);
        atomicAdd(&gsq[m0 + rr], s2);
    }
}

// ---------------------------------------------------------------------------
// Kernel 5: BatchNorm (training stats) + affine + ReLU + residual.
// ---------------------------------------------------------------------------
__global__ __launch_bounds__(256)
void bn_relu_kernel(const float* __restrict__ x,
                    const float* __restrict__ gamma,
                    const float* __restrict__ beta,
                    const float* __restrict__ yy,
                    const float* __restrict__ gsum,
                    const float* __restrict__ gsq,
                    float* __restrict__ out) {
    const int idx = blockIdx.x * 256 + threadIdx.x;
    if (idx >= NB * NC * NP) return;
    const int c = (idx / NP) % NC;
    const float inv  = 1.0f / (float)(NB * NP);
    const float mean = gsum[c] * inv;
    const float var  = gsq[c] * inv - mean * mean;
    float yv = (yy[idx] - mean) * rsqrtf(var + BN_EPS);
    yv = yv * gamma[c] + beta[c];
    out[idx] = x[idx] + fmaxf(yv, 0.0f);
}

// ---------------------------------------------------------------------------
// Launcher
// ---------------------------------------------------------------------------
extern "C" void kernel_launch(void* const* d_in, const int* in_sizes, int n_in,
                              void* d_out, int out_size, void* d_ws, size_t ws_size,
                              hipStream_t stream) {
    const float* x     = (const float*)d_in[0];
    const float* wq    = (const float*)d_in[1];
    const float* wk    = (const float*)d_in[2];
    const float* wv    = (const float*)d_in[3];
    const float* wt    = (const float*)d_in[4];
    const float* bt    = (const float*)d_in[5];
    const float* gamma = (const float*)d_in[6];
    const float* beta  = (const float*)d_in[7];
    float* out = (float*)d_out;

    // Workspace layout
    constexpr size_t ELEMS = (size_t)NB * NC * NP;   // 4,194,304
    constexpr size_t WELEMS = (size_t)NC * NC;       // 65,536
    char* ws = (char*)d_ws;
    _Float16* xh   = (_Float16*)(ws);                        //  8 MB
    _Float16* qT   = (_Float16*)(ws + ELEMS * 2);            //  8 MB
    _Float16* kT   = (_Float16*)(ws + ELEMS * 4);            //  8 MB
    _Float16* vv   = (_Float16*)(ws + ELEMS * 6);            //  8 MB
    _Float16* dhp  = (_Float16*)(ws + ELEMS * 8);            //  8 MB
    float*    yy   = (float*)   (ws + ELEMS * 10);           // 16 MB
    char*     wbase = ws + ELEMS * 14;
    _Float16* wqh  = (_Float16*)(wbase);
    _Float16* wkh  = (_Float16*)(wbase + WELEMS * 2);
    _Float16* wvh  = (_Float16*)(wbase + WELEMS * 4);
    _Float16* wth  = (_Float16*)(wbase + WELEMS * 6);
    float*    gsum = (float*)   (wbase + WELEMS * 8);
    float*    gsq  = (float*)   (wbase + WELEMS * 8 + 4096);

    zero_stats_kernel<<<1, 256, 0, stream>>>(gsum, gsq);
    cvt_weights_kernel<<<(NC * NC + 255) / 256, 256, 0, stream>>>(
        wq, wk, wv, wt, wqh, wkh, wvh, wth);
    cvt_x_kernel<<<(int)((ELEMS + 255) / 256), 256, 0, stream>>>(x, xh);

    dim3 gemm_grid(NP / 128, NC / 16, NB);
    proj_qkv_kernel<<<gemm_grid, 256, 0, stream>>>(xh, wqh, wkh, wvh, qT, kT, vv);

    dim3 attn_grid(NP / 16, NH, NB);
    flash_attn_kernel<<<attn_grid, 32, 0, stream>>>(qT, kT, vv, x, dhp);

    trans_conv_kernel<<<gemm_grid, 256, 0, stream>>>(dhp, wth, bt, yy, gsum, gsq);

    const int n = NB * NC * NP;
    bn_relu_kernel<<<(n + 255) / 256, 256, 0, stream>>>(x, gamma, beta, yy, gsum, gsq, out);
}
